// ProductKeyMemory_30597347017054
// MI455X (gfx1250) — matrix operand order repaired and checked by reading, not compiled, not run on hardware
//
#include <hip/hip_runtime.h>

typedef __bf16 bf16;
typedef __attribute__((ext_vector_type(16))) __bf16          v16bf;
typedef __attribute__((ext_vector_type(8)))  float           v8f;
typedef __attribute__((ext_vector_type(8)))  unsigned short  v8us;

#define N_TOK 8192
#define QDIM  1024
#define HALFD 512
#define CB    1024
#define KC    64
#define TAKE  32
#define VDIM  512
#define NEG_INF (-3.402823466e38f)

// LDS tile geometry for the GEMM (per buffer): A 128x32 bf16 + B 64x32 bf16
#define TILE_A_US (128 * 32)          // 4096 ushorts = 8 KB
#define TILE_B_US (64 * 32)           // 2048 ushorts = 4 KB
#define TILE_US   (TILE_A_US + TILE_B_US)

// ---------------------------------------------------------------------------
// fp32 -> bf16 (round-to-nearest-even)
// ---------------------------------------------------------------------------
__device__ __forceinline__ bf16 f2bf(float x) {
  unsigned u = __builtin_bit_cast(unsigned, x);
  unsigned r = (u + 0x7FFFu + ((u >> 16) & 1u)) >> 16;
  unsigned short s = (unsigned short)r;
  return __builtin_bit_cast(bf16, s);
}

__global__ void cvt_f32_bf16(const float* __restrict__ in, bf16* __restrict__ out, int n) {
  int i = blockIdx.x * blockDim.x + threadIdx.x;
  int stride = gridDim.x * blockDim.x;
  for (; i < n; i += stride) out[i] = f2bf(in[i]);
}

// Wt[n*K + k] = W[k*N + n]  (K-contiguous B operand)
__global__ void transpose_cvt(const float* __restrict__ W, bf16* __restrict__ Wt, int K, int N) {
  int i = blockIdx.x * blockDim.x + threadIdx.x;
  int total = K * N;
  int stride = gridDim.x * blockDim.x;
  for (; i < total; i += stride) {
    int k = i / N, n = i % N;
    Wt[(size_t)n * K + k] = f2bf(W[i]);
  }
}

// ---------------------------------------------------------------------------
// Async Global->LDS copy (gfx1250): per-lane 16B transfer, tracked by ASYNCcnt
// ---------------------------------------------------------------------------
__device__ __forceinline__ void async_copy_b128(unsigned lds_addr, unsigned long long gaddr) {
  asm volatile("global_load_async_to_lds_b128 %0, %1, off"
               :: "v"(lds_addr), "v"(gaddr) : "memory");
}
__device__ __forceinline__ void wait_async0() {
  asm volatile("s_wait_asynccnt 0x0" ::: "memory");
}

// ---------------------------------------------------------------------------
// WMMA fragment reads from the LDS tile (bf16 16x16x32, wave32 layouts).
// Tile rows are K-contiguous (32 elems, 64 B) so per-lane reads are b128s.
// ---------------------------------------------------------------------------
__device__ __forceinline__ v16bf frag_a_lds(const unsigned short* buf, int row0, int lane) {
  int m = lane & 15;
  int h = (lane >> 4) << 3;                       // 0 or 8
  const unsigned short* base = buf + (row0 + m) * 32 + h;
  union { v16bf v; v8us u[2]; } r;
  r.u[0] = *reinterpret_cast<const v8us*>(base);        // K = h .. h+7
  r.u[1] = *reinterpret_cast<const v8us*>(base + 16);   // K = 16+h .. 16+h+7
  return r.v;
}
__device__ __forceinline__ v16bf frag_b_lds(const unsigned short* buf, int col0, int lane) {
  int n  = lane & 15;
  int ko = (lane >> 4) << 3;                      // 0 or 8
  const unsigned short* base = buf + TILE_A_US + (col0 + n) * 32 + ko;
  union { v16bf v; v8us u[2]; } r;
  r.u[0] = *reinterpret_cast<const v8us*>(base);
  r.u[1] = *reinterpret_cast<const v8us*>(base + 16);
  return r.v;
}

// ---------------------------------------------------------------------------
// GEMM: C[M,N] = A[M,K](bf16,row-major,+aoff) @ Bt[N,K]^T (bf16)
// Block tile 128x64 (8 wave32s, wave tile 32x32), K-step 32.
// Double-buffered LDS staging via async Global->LDS copies:
//   each block loads every tile element exactly once (12 KB/step) while the
//   previous step's WMMAs execute; s_wait_asynccnt joins a full stage later.
// ---------------------------------------------------------------------------
__global__ __launch_bounds__(256) void gemm_bf16_wmma(
    const bf16* __restrict__ A, int lda, int aoff,
    const bf16* __restrict__ Bt, int ldb,
    const float* __restrict__ bias,
    float* __restrict__ Cf, bf16* __restrict__ Cb, int ldc, int K)
{
  __shared__ __align__(16) unsigned short tiles[2][TILE_US];

  int tid  = threadIdx.x;
  int lane = tid & 31;
  int wave = tid >> 5;
  int wm = wave & 3, wn = wave >> 2;
  int mB = blockIdx.x * 128;
  int nB = blockIdx.y * 64;
  int m0w = wm * 32;                  // wave tile origin within block tile
  int n0w = wn * 32;

  const bf16* Abase = A + aoff;
  unsigned lbase = (unsigned)(size_t)(&tiles[0][0]);

  // thread -> 16B chunks: A gets 2 chunks/thread (512 total), B gets 1 (256)
  int ra0 = (tid)        >> 2, ca0 = (tid)        & 3;
  int ra1 = (tid + 256)  >> 2, ca1 = (tid + 256)  & 3;
  int rb  = tid >> 2,          cbk = tid & 3;

  auto issue_tiles = [&](int buf, int k0) {
    unsigned lb = lbase + (unsigned)buf * (TILE_US * 2);
    async_copy_b128(lb + (unsigned)(ra0 * 32 + ca0 * 8) * 2,
                    (unsigned long long)(size_t)(Abase + (size_t)(mB + ra0) * lda + k0 + ca0 * 8));
    async_copy_b128(lb + (unsigned)(ra1 * 32 + ca1 * 8) * 2,
                    (unsigned long long)(size_t)(Abase + (size_t)(mB + ra1) * lda + k0 + ca1 * 8));
    async_copy_b128(lb + (unsigned)(TILE_A_US + rb * 32 + cbk * 8) * 2,
                    (unsigned long long)(size_t)(Bt + (size_t)(nB + rb) * ldb + k0 + cbk * 8));
  };

  v8f c00 = {}, c01 = {}, c10 = {}, c11 = {};

  issue_tiles(0, 0);                  // prologue stage
  int buf = 0;
  for (int k0 = 0; k0 < K; k0 += 32) {
    wait_async0();                    // joins copies issued one stage earlier
    __syncthreads();
    if (k0 + 32 < K) issue_tiles(buf ^ 1, k0 + 32);

    const unsigned short* bp = &tiles[buf][0];
    v16bf a0 = frag_a_lds(bp, m0w,      lane);
    v16bf a1 = frag_a_lds(bp, m0w + 16, lane);
    v16bf b0 = frag_b_lds(bp, n0w,      lane);
    v16bf b1 = frag_b_lds(bp, n0w + 16, lane);
    c00 = __builtin_amdgcn_wmma_f32_16x16x32_bf16(false, a0, false, b0, (short)0, c00, false, false);
    c01 = __builtin_amdgcn_wmma_f32_16x16x32_bf16(false, a0, false, b1, (short)0, c01, false, false);
    c10 = __builtin_amdgcn_wmma_f32_16x16x32_bf16(false, a1, false, b0, (short)0, c10, false, false);
    c11 = __builtin_amdgcn_wmma_f32_16x16x32_bf16(false, a1, false, b1, (short)0, c11, false, false);

    __syncthreads();                  // all waves done reading buf before reuse
    buf ^= 1;
  }

  // C/D layout: lanes 0-15 hold rows r, lanes 16-31 rows r+8; col = lane&15
  int col = lane & 15;
  int rhi = (lane >> 4) * 8;
  int m0 = mB + m0w, n0 = nB + n0w;
  int cA = n0 + col, cBc = n0 + 16 + col;
  float bA = bias ? bias[cA]  : 0.f;
  float bB = bias ? bias[cBc] : 0.f;
#pragma unroll
  for (int r = 0; r < 8; ++r) {
    int row0 = m0 + rhi + r;
    float v00 = c00[r] + bA, v01 = c01[r] + bB;
    float v10 = c10[r] + bA, v11 = c11[r] + bB;
    if (Cf) {
      Cf[(size_t)row0 * ldc + cA]         = v00;
      Cf[(size_t)row0 * ldc + cBc]        = v01;
      Cf[(size_t)(row0 + 16) * ldc + cA]  = v10;
      Cf[(size_t)(row0 + 16) * ldc + cBc] = v11;
    } else {
      Cb[(size_t)row0 * ldc + cA]         = f2bf(v00);
      Cb[(size_t)row0 * ldc + cBc]        = f2bf(v01);
      Cb[(size_t)(row0 + 16) * ldc + cA]  = f2bf(v10);
      Cb[(size_t)(row0 + 16) * ldc + cBc] = f2bf(v11);
    }
  }
}

// ---------------------------------------------------------------------------
// Per-row top-64 (value + index, descending, lower index wins ties)
// ---------------------------------------------------------------------------
__global__ __launch_bounds__(256) void topk64_kernel(const float* __restrict__ score,
                                                     float* __restrict__ tops,
                                                     int* __restrict__ topi)
{
  __shared__ float sv[CB];
  __shared__ float rv[256];
  __shared__ int   ri[256];
  int row = blockIdx.x;
  int tid = threadIdx.x;
  const float* srow = score + (size_t)row * CB;
  for (int i = tid; i < CB; i += 256) sv[i] = srow[i];
  __syncthreads();
  for (int t = 0; t < KC; ++t) {
    float bv = NEG_INF; int bi = -1;
    for (int i = tid; i < CB; i += 256) {
      float v = sv[i];
      if (v > bv) { bv = v; bi = i; }
    }
    rv[tid] = bv; ri[tid] = bi;
    __syncthreads();
    for (int s = 128; s > 0; s >>= 1) {
      if (tid < s) {
        bool take = (rv[tid + s] > rv[tid]) ||
                    (rv[tid + s] == rv[tid] && ri[tid + s] >= 0 &&
                     (ri[tid] < 0 || ri[tid + s] < ri[tid]));
        if (take) { rv[tid] = rv[tid + s]; ri[tid] = ri[tid + s]; }
      }
      __syncthreads();
    }
    if (tid == 0) {
      tops[(size_t)row * KC + t] = rv[0];
      topi[(size_t)row * KC + t] = ri[0];
      sv[ri[0]] = NEG_INF;
    }
    __syncthreads();
  }
}

// ---------------------------------------------------------------------------
// Per-row: top-32 over 64x64 cartesian pair scores, softmax, weighted gather.
// ---------------------------------------------------------------------------
__global__ __launch_bounds__(256) void pair_topk_gather_kernel(
    const float* __restrict__ tops_a, const int* __restrict__ idx_a,
    const float* __restrict__ tops_b, const int* __restrict__ idx_b,
    const float* __restrict__ values, const float* __restrict__ value_scale,
    float* __restrict__ out_w, int* __restrict__ out_idx, float* __restrict__ out_s)
{
  __shared__ float sa[KC], sb[KC];
  __shared__ int   ia[KC], ib[KC];
  __shared__ unsigned mask[KC * KC / 32];   // 4096-bit exclusion mask
  __shared__ float rv[256];
  __shared__ int   ri[256];
  __shared__ float wts[TAKE];
  __shared__ int   gidx[TAKE];

  int row = blockIdx.x, tid = threadIdx.x;
  if (tid < KC) {
    sa[tid] = tops_a[(size_t)row * KC + tid];
    sb[tid] = tops_b[(size_t)row * KC + tid];
    ia[tid] = idx_a[(size_t)row * KC + tid];
    ib[tid] = idx_b[(size_t)row * KC + tid];
  }
  if (tid < KC * KC / 32) mask[tid] = 0u;
  __syncthreads();

  for (int t = 0; t < TAKE; ++t) {
    float bv = NEG_INF; int bp = -1;
    for (int p = tid; p < KC * KC; p += 256) {
      if (mask[p >> 5] & (1u << (p & 31))) continue;
      float v = sa[p >> 6] + sb[p & 63];
      if (v > bv) { bv = v; bp = p; }
    }
    rv[tid] = bv; ri[tid] = bp;
    __syncthreads();
    for (int s = 128; s > 0; s >>= 1) {
      if (tid < s) {
        bool take = (rv[tid + s] > rv[tid]) ||
                    (rv[tid + s] == rv[tid] && ri[tid + s] >= 0 &&
                     (ri[tid] < 0 || ri[tid + s] < ri[tid]));
        if (take) { rv[tid] = rv[tid + s]; ri[tid] = ri[tid + s]; }
      }
      __syncthreads();
    }
    if (tid == 0) {
      int p = ri[0];
      mask[p >> 5] |= (1u << (p & 31));
      int g = ia[p >> 6] * CB + ib[p & 63];
      gidx[t] = g;
      wts[t]  = rv[0];
      out_s[(size_t)row * TAKE + t]   = rv[0];
      out_idx[(size_t)row * TAKE + t] = g;
    }
    __syncthreads();
  }

  // prefetch the 32 winning value rows (32 x 2 KB) while softmax runs:
  // 256 lanes x 256 B apart covers every cacheline of every selected row
  {
    int t   = tid >> 3;               // 0..31 slot
    int seg = tid & 7;                // 8 x 256 B segments per 2 KB row
    __builtin_prefetch(values + (size_t)gidx[t] * VDIM + seg * 64, 0, 3);
  }

  if (tid == 0) {
    float m = wts[0];                 // descending -> max first
    float s = 0.f;
    for (int t = 0; t < TAKE; ++t) { wts[t] = __expf(wts[t] - m); s += wts[t]; }
    float inv = 1.f / s;
    for (int t = 0; t < TAKE; ++t) wts[t] *= inv;
  }
  __syncthreads();

  float vs = *value_scale;
  for (int d = tid; d < VDIM; d += 256) {   // coalesced 2 KB row reads x 32 slots
    float acc = 0.f;
#pragma unroll 4
    for (int t = 0; t < TAKE; ++t)
      acc += wts[t] * values[(size_t)gidx[t] * VDIM + d];
    out_w[(size_t)row * VDIM + d] = acc * vs;
  }
}

// ---------------------------------------------------------------------------
extern "C" void kernel_launch(void* const* d_in, const int* in_sizes, int n_in,
                              void* d_out, int out_size, void* d_ws, size_t ws_size,
                              hipStream_t stream)
{
  const float* queries = (const float*)d_in[0];   // [8192,1024]
  const float* Wq      = (const float*)d_in[1];   // [1024,1024]
  const float* bq      = (const float*)d_in[2];   // [1024]
  const float* ca      = (const float*)d_in[3];   // [1024,512]
  const float* cb      = (const float*)d_in[4];   // [1024,512]
  const float* values  = (const float*)d_in[5];   // [1048576,512]
  const float* vscale  = (const float*)d_in[6];   // scalar
  (void)in_sizes; (void)n_in; (void)out_size; (void)ws_size;

  char* ws = (char*)d_ws;
  size_t off = 0;
  auto alloc = [&](size_t bytes) -> char* {
    off = (off + 255) & ~(size_t)255;
    char* p = ws + off;
    off += bytes;
    return p;
  };

  bf16*  qin_bf  = (bf16*)alloc((size_t)N_TOK * QDIM * 2);
  bf16*  WqT     = (bf16*)alloc((size_t)QDIM * QDIM * 2);
  bf16*  ca_bf   = (bf16*)alloc((size_t)CB * HALFD * 2);
  bf16*  cb_bf   = (bf16*)alloc((size_t)CB * HALFD * 2);
  bf16*  q_bf    = (bf16*)alloc((size_t)N_TOK * QDIM * 2);
  float* score_a = (float*)alloc((size_t)N_TOK * CB * 4);
  float* score_b = (float*)alloc((size_t)N_TOK * CB * 4);
  float* tops_a  = (float*)alloc((size_t)N_TOK * KC * 4);
  int*   idxw_a  = (int*)  alloc((size_t)N_TOK * KC * 4);
  float* tops_b  = (float*)alloc((size_t)N_TOK * KC * 4);
  int*   idxw_b  = (int*)  alloc((size_t)N_TOK * KC * 4);

  // 1) precision prep
  cvt_f32_bf16<<<1024, 256, 0, stream>>>(queries, qin_bf, N_TOK * QDIM);
  cvt_f32_bf16<<<256,  256, 0, stream>>>(ca, ca_bf, CB * HALFD);
  cvt_f32_bf16<<<256,  256, 0, stream>>>(cb, cb_bf, CB * HALFD);
  transpose_cvt<<<1024, 256, 0, stream>>>(Wq, WqT, QDIM, QDIM);

  // 2) projection GEMM (async-LDS staged bf16 WMMA, fp32 acc, bias, bf16 out)
  dim3 g1(N_TOK / 128, QDIM / 64);
  gemm_bf16_wmma<<<g1, 256, 0, stream>>>(qin_bf, QDIM, 0, WqT, QDIM, bq,
                                         nullptr, q_bf, QDIM, QDIM);

  // 3) score GEMMs: qa @ ca^T, qb @ cb^T
  dim3 g2(N_TOK / 128, CB / 64);
  gemm_bf16_wmma<<<g2, 256, 0, stream>>>(q_bf, QDIM, 0,     ca_bf, HALFD, nullptr,
                                         score_a, nullptr, CB, HALFD);
  gemm_bf16_wmma<<<g2, 256, 0, stream>>>(q_bf, QDIM, HALFD, cb_bf, HALFD, nullptr,
                                         score_b, nullptr, CB, HALFD);

  // 4) per-row top-64 of each score matrix
  topk64_kernel<<<N_TOK, 256, 0, stream>>>(score_a, tops_a, idxw_a);
  topk64_kernel<<<N_TOK, 256, 0, stream>>>(score_b, tops_b, idxw_b);

  // 5) pair top-32 + softmax + weighted gather (bandwidth-dominant stage)
  float* out_w = (float*)d_out;                                   // [8192,512]
  int*   out_i = (int*)(out_w + (size_t)N_TOK * VDIM);            // [8192,32]
  float* out_s = (float*)(out_w + (size_t)N_TOK * VDIM + (size_t)N_TOK * TAKE);
  pair_topk_gather_kernel<<<N_TOK, 256, 0, stream>>>(tops_a, idxw_a, tops_b, idxw_b,
                                                     values, vscale,
                                                     out_w, out_i, out_s);
}
